// S5_14070312861752
// MI455X (gfx1250) — matrix operand order repaired and checked by reading, not compiled, not run on hardware
//
#include <hip/hip_runtime.h>
#include <hip/hip_bf16.h>

// S5 bidirectional SSM layer for MI455X (gfx1250), bf16 WMMA pipeline:
//   1) prep: Lambda_bar, W1 = [Re(B_bar); Im(B_bar)] bf16, W2 = [2C_re | -2C_im | ...] bf16
//   2) GEMM1 (WMMA bf16): Bu(L,2P) = u @ W1^T
//   3) chunked complex scan (fwd+bwd), lane = channel for coalescing -> X(L,4P) bf16
//   4) GEMM2 (WMMA bf16): ys = X @ W2^T + D*u

#define L_DIM 16384
#define H_DIM 1024
#define P_DIM 512
#define K2_DIM (4 * P_DIM)   // 2048

typedef __attribute__((ext_vector_type(16))) __bf16 v16bf;
typedef __attribute__((ext_vector_type(8)))  __bf16 v8bf;
typedef __attribute__((ext_vector_type(4)))  __bf16 v4bf;
typedef __attribute__((ext_vector_type(8)))  float  v8f;
typedef __attribute__((ext_vector_type(4)))  float  v4f;

union FragBF { v16bf v; v8bf h[2]; };

// ---------------------------------------------------------------- prep kernels

__global__ void s5_prep_lambda(const float* __restrict__ Lre,
                               const float* __restrict__ Lim,
                               const float* __restrict__ lstep,
                               float* __restrict__ LamBar /* (P,2) */) {
    int p = blockIdx.x * blockDim.x + threadIdx.x;
    if (p >= P_DIM) return;
    float st = __expf(lstep[p]);
    float er = __expf(st * Lre[p]);
    LamBar[2 * p + 0] = er * __cosf(st * Lim[p]);
    LamBar[2 * p + 1] = er * __sinf(st * Lim[p]);
}

__global__ void s5_build_W1(const float* __restrict__ Lre,
                            const float* __restrict__ Lim,
                            const float* __restrict__ Bri,
                            const float* __restrict__ lstep,
                            __bf16* __restrict__ W1 /* (2P, H) row-major */) {
    int idx = blockIdx.x * blockDim.x + threadIdx.x;   // P*H threads
    if (idx >= P_DIM * H_DIM) return;
    int p = idx >> 10;            // / H_DIM
    int h = idx & (H_DIM - 1);
    float st  = __expf(lstep[p]);
    float lre = Lre[p], lim = Lim[p];
    float er  = __expf(st * lre);
    float lbr = er * __cosf(st * lim);
    float lbi = er * __sinf(st * lim);
    // c = (Lambda_bar - 1) / Lambda
    float nr = lbr - 1.0f, ni = lbi;
    float inv = 1.0f / (lre * lre + lim * lim);
    float cr = (nr * lre + ni * lim) * inv;
    float ci = (ni * lre - nr * lim) * inv;
    float br = Bri[(size_t)idx * 2 + 0];
    float bi = Bri[(size_t)idx * 2 + 1];
    W1[(size_t)p * H_DIM + h]           = (__bf16)(cr * br - ci * bi);
    W1[(size_t)(P_DIM + p) * H_DIM + h] = (__bf16)(cr * bi + ci * br);
}

__global__ void s5_build_W2(const float* __restrict__ Cri,
                            __bf16* __restrict__ W2 /* (H, 4P) row-major */) {
    int idx = blockIdx.x * blockDim.x + threadIdx.x;   // H*4P threads
    if (idx >= H_DIM * K2_DIM) return;
    int h = idx >> 11;            // / K2_DIM
    int j = idx & (K2_DIM - 1);
    int blk = j >> 9;             // 0:C1re 1:C1im 2:C2re 3:C2im
    int q   = j & (P_DIM - 1);
    int jj  = (blk & 2) ? (P_DIM + q) : q;
    int ri  = blk & 1;
    float sgn = ri ? -2.0f : 2.0f;
    float c = Cri[((size_t)h * (2 * P_DIM) + jj) * 2 + ri];
    W2[(size_t)idx] = (__bf16)(sgn * c);
}

__global__ void s5_cvt_bf16(const float* __restrict__ src,
                            __bf16* __restrict__ dst, int n4) {
    int i = blockIdx.x * blockDim.x + threadIdx.x;
    if (i >= n4) return;
    v4f x = *(const v4f*)(src + (size_t)i * 4);
    v4bf y;
    y[0] = (__bf16)x[0]; y[1] = (__bf16)x[1];
    y[2] = (__bf16)x[2]; y[3] = (__bf16)x[3];
    *(v4bf*)(dst + (size_t)i * 4) = y;
}

// ------------------------------------------------------------- WMMA bf16 GEMM
// C(MxN,f32) = A(MxK,bf16 rowmajor) @ Bt(NxK,bf16 rowmajor)^T  [+ D[n]*U[m,n]]
// Block: 128 threads = 4 waves; wave w -> rows [m0, m0+32); block -> cols [n0, n0+64).
// Per wave: 2 A fragments x 4 B fragments = 8 WMMAs per K-step (32x64 tile).

__global__ void __launch_bounds__(128)
s5_gemm_wmma(const __bf16* __restrict__ A,
             const __bf16* __restrict__ Bt,
             float* __restrict__ C,
             int M, int N, int K,
             const float* __restrict__ D,
             const float* __restrict__ U) {
    const int lane = threadIdx.x & 31;
    const int wave = threadIdx.x >> 5;
    const int m0 = blockIdx.y * 128 + wave * 32;
    const int n0 = blockIdx.x * 64;

    const int hi  = lane >> 4;        // 0 | 1
    const int l15 = lane & 15;
    const int ka  = hi * 16;          // A: K {0-15} | {16-31}
    const int kb  = hi * 16;          // B: K {0-15} | {16-31}

    const __bf16* arow0 = A + (size_t)(m0 + l15) * K + ka;
    const __bf16* arow1 = arow0 + (size_t)16 * K;
    const __bf16* brow0 = Bt + (size_t)(n0 + 0 * 16 + l15) * K + kb;
    const __bf16* brow1 = Bt + (size_t)(n0 + 1 * 16 + l15) * K + kb;
    const __bf16* brow2 = Bt + (size_t)(n0 + 2 * 16 + l15) * K + kb;
    const __bf16* brow3 = Bt + (size_t)(n0 + 3 * 16 + l15) * K + kb;

    v8f acc[8] = {};

    for (int k0 = 0; k0 < K; k0 += 32) {
        __builtin_prefetch(arow0 + k0 + 256, 0, 0);   // global_prefetch_b8
        __builtin_prefetch(arow1 + k0 + 256, 0, 0);
        FragBF a0, a1;
        a0.h[0] = *(const v8bf*)(arow0 + k0);
        a0.h[1] = *(const v8bf*)(arow0 + k0 + 8);
        a1.h[0] = *(const v8bf*)(arow1 + k0);
        a1.h[1] = *(const v8bf*)(arow1 + k0 + 8);
        FragBF b0, b1, b2, b3;
        b0.h[0] = *(const v8bf*)(brow0 + k0);
        b0.h[1] = *(const v8bf*)(brow0 + k0 + 8);
        b1.h[0] = *(const v8bf*)(brow1 + k0);
        b1.h[1] = *(const v8bf*)(brow1 + k0 + 8);
        b2.h[0] = *(const v8bf*)(brow2 + k0);
        b2.h[1] = *(const v8bf*)(brow2 + k0 + 8);
        b3.h[0] = *(const v8bf*)(brow3 + k0);
        b3.h[1] = *(const v8bf*)(brow3 + k0 + 8);
        acc[0] = __builtin_amdgcn_wmma_f32_16x16x32_bf16(false, a0.v, false, b0.v,
                                                         (short)0, acc[0], false, false);
        acc[4] = __builtin_amdgcn_wmma_f32_16x16x32_bf16(false, a1.v, false, b0.v,
                                                         (short)0, acc[4], false, false);
        acc[1] = __builtin_amdgcn_wmma_f32_16x16x32_bf16(false, a0.v, false, b1.v,
                                                         (short)0, acc[1], false, false);
        acc[5] = __builtin_amdgcn_wmma_f32_16x16x32_bf16(false, a1.v, false, b1.v,
                                                         (short)0, acc[5], false, false);
        acc[2] = __builtin_amdgcn_wmma_f32_16x16x32_bf16(false, a0.v, false, b2.v,
                                                         (short)0, acc[2], false, false);
        acc[6] = __builtin_amdgcn_wmma_f32_16x16x32_bf16(false, a1.v, false, b2.v,
                                                         (short)0, acc[6], false, false);
        acc[3] = __builtin_amdgcn_wmma_f32_16x16x32_bf16(false, a0.v, false, b3.v,
                                                         (short)0, acc[3], false, false);
        acc[7] = __builtin_amdgcn_wmma_f32_16x16x32_bf16(false, a1.v, false, b3.v,
                                                         (short)0, acc[7], false, false);
    }

    // D-matrix layout: VGPR v <-> M = v + 8*hi, N = lane&15
    const int mofs = hi * 8;
#pragma unroll
    for (int j = 0; j < 4; ++j) {
        const int n = n0 + j * 16 + l15;
#pragma unroll
        for (int half = 0; half < 2; ++half) {
            const int mbase = m0 + half * 16 + mofs;
#pragma unroll
            for (int v = 0; v < 8; ++v) {
                const size_t idx = (size_t)(mbase + v) * N + n;
                float val = acc[half * 4 + j][v];
                if (D) val += D[n] * U[idx];
                C[idx] = val;
            }
        }
    }
}

// ------------------------------------------------------- chunked complex scan
// Bu: (L, 2P) f32 [re | im]; X: (L, 4P) bf16 [fwd_re | fwd_im | bwd_re | bwd_im]
// Block = (32 adjacent channels) x (one direction), 1024 threads:
//   lane c = channel-in-group (coalesced 128B loads), s = time segment (32 x 512).
// Segment carries scanned per-channel through LDS with multiplier lambda^(512*2^s).

__global__ void __launch_bounds__(1024)
s5_scan(const float* __restrict__ Bu,
        const float* __restrict__ LamBar,
        __bf16* __restrict__ X, int Ldim) {
    const int c   = threadIdx.x & 31;       // channel within group
    const int s   = threadIdx.x >> 5;       // segment 0..31
    const int p   = blockIdx.x * 32 + c;
    const int dir = blockIdx.y;             // 0 fwd, 1 bwd
    const int NSEG = 32;
    const int SEG  = Ldim / NSEG;           // 512

    const float lr = LamBar[2 * p + 0];
    const float li = LamBar[2 * p + 1];

    __shared__ float sr[32][32], si[32][32];

    // pass 1: local scan with zero carry -> segment end value
    float cr = 0.0f, ci = 0.0f;
    const int base = s * SEG;
    for (int k = 0; k < SEG; ++k) {
        int t = dir ? (Ldim - 1 - (base + k)) : (base + k);
        float br = Bu[(size_t)t * (2 * P_DIM) + p];
        float bi = Bu[(size_t)t * (2 * P_DIM) + P_DIM + p];
        float nr = lr * cr - li * ci + br;
        ci = lr * ci + li * cr + bi;
        cr = nr;
    }
    sr[s][c] = cr;
    si[s][c] = ci;

    // f = lambda^SEG  (SEG = 512 -> 9 squarings)
    float fr = lr, fi = li;
    for (int q = 0; q < 9; ++q) {
        float t0 = fr * fr - fi * fi;
        fi = 2.0f * fr * fi;
        fr = t0;
    }

    // inclusive scan of segment carries (per channel) across 32 segments
    for (int off = 1; off < NSEG; off <<= 1) {
        __syncthreads();
        float pr = 0.0f, pi = 0.0f;
        if (s >= off) { pr = sr[s - off][c]; pi = si[s - off][c]; }
        __syncthreads();
        if (s >= off) {
            cr += fr * pr - fi * pi;
            ci += fr * pi + fi * pr;
            sr[s][c] = cr;
            si[s][c] = ci;
        }
        float t0 = fr * fr - fi * fi;   // f <- f^2
        fi = 2.0f * fr * fi;
        fr = t0;
    }
    __syncthreads();
    float xr = 0.0f, xi = 0.0f;
    if (s > 0) { xr = sr[s - 1][c]; xi = si[s - 1][c]; }   // exclusive carry-in

    // pass 2: re-run recurrence from carry and emit bf16
    const int ob = dir ? (2 * P_DIM) : 0;
    for (int k = 0; k < SEG; ++k) {
        int t = dir ? (Ldim - 1 - (base + k)) : (base + k);
        float br = Bu[(size_t)t * (2 * P_DIM) + p];
        float bi = Bu[(size_t)t * (2 * P_DIM) + P_DIM + p];
        float nr = lr * xr - li * xi + br;
        xi = lr * xi + li * xr + bi;
        xr = nr;
        X[(size_t)t * K2_DIM + ob + p]         = (__bf16)xr;
        X[(size_t)t * K2_DIM + ob + P_DIM + p] = (__bf16)xi;
    }
}

// -------------------------------------------------------------------- launch

extern "C" void kernel_launch(void* const* d_in, const int* in_sizes, int n_in,
                              void* d_out, int out_size, void* d_ws, size_t ws_size,
                              hipStream_t stream) {
    (void)in_sizes; (void)n_in; (void)out_size; (void)ws_size;
    const float* u     = (const float*)d_in[0];   // (L,H)
    const float* Lre   = (const float*)d_in[1];   // (P,)
    const float* Lim   = (const float*)d_in[2];   // (P,)
    const float* Bri   = (const float*)d_in[3];   // (P,H,2)
    const float* Cri   = (const float*)d_in[4];   // (H,2P,2)
    const float* Dv    = (const float*)d_in[5];   // (H,)
    const float* lstep = (const float*)d_in[6];   // (P,)
    float* out = (float*)d_out;                   // (L,H)

    char* ws = (char*)d_ws;
    size_t off = 0;
    auto take = [&](size_t bytes) -> char* {
        char* ptr = ws + off;
        off = (off + bytes + 255) & ~(size_t)255;
        return ptr;
    };
    float*  LamBar = (float*)take((size_t)P_DIM * 2 * sizeof(float));
    __bf16* W1     = (__bf16*)take((size_t)2 * P_DIM * H_DIM * sizeof(__bf16));
    __bf16* W2     = (__bf16*)take((size_t)H_DIM * K2_DIM * sizeof(__bf16));
    __bf16* Ub     = (__bf16*)take((size_t)L_DIM * H_DIM * sizeof(__bf16));
    float*  Y1     = (float*)take((size_t)L_DIM * 2 * P_DIM * sizeof(float));
    __bf16* X      = (__bf16*)take((size_t)L_DIM * K2_DIM * sizeof(__bf16));

    // 1. discretization / weight folding
    s5_prep_lambda<<<dim3((P_DIM + 255) / 256), dim3(256), 0, stream>>>(Lre, Lim, lstep, LamBar);
    s5_build_W1<<<dim3((P_DIM * H_DIM) / 256), dim3(256), 0, stream>>>(Lre, Lim, Bri, lstep, W1);
    s5_build_W2<<<dim3((H_DIM * K2_DIM) / 256), dim3(256), 0, stream>>>(Cri, W2);
    s5_cvt_bf16<<<dim3((L_DIM * H_DIM / 4) / 256), dim3(256), 0, stream>>>(u, Ub, L_DIM * H_DIM / 4);

    // 2. Bu = u @ B_bar^T  (one real GEMM, N = 2P)
    s5_gemm_wmma<<<dim3((2 * P_DIM) / 64, L_DIM / 128), dim3(128), 0, stream>>>(
        Ub, W1, Y1, L_DIM, 2 * P_DIM, H_DIM, nullptr, nullptr);

    // 3. bidirectional scan (coalesced: lane = channel)
    s5_scan<<<dim3(P_DIM / 32, 2), dim3(1024), 0, stream>>>(Y1, LamBar, X, L_DIM);

    // 4. ys = X @ W2^T + D*u  (K = 4P)
    s5_gemm_wmma<<<dim3(H_DIM / 64, L_DIM / 128), dim3(128), 0, stream>>>(
        X, W2, out, L_DIM, H_DIM, K2_DIM, Dv, u);
}
